// GCN_MUTAG_36644660969782
// MI455X (gfx1250) — compile-verified, matrix operand order checked by reading, not run on hardware
//
#include <hip/hip_runtime.h>
#include <hip/hip_fp16.h>

typedef __attribute__((ext_vector_type(16))) _Float16 v16h;
typedef __attribute__((ext_vector_type(8)))  _Float16 v8h;
typedef __attribute__((ext_vector_type(8)))  float    v8f;

#define N_NODES  100000
#define N_EDGES  3200000
#define N_GRAPHS 1024
#define STRIDE   64   // padded feature stride for all node buffers
#define M_TILES  6250 // 100000 / 16 exactly

// ---------------------------------------------------------------- utilities

__global__ void zero_f32_kernel(float* p, int n) {
    int t = blockIdx.x * blockDim.x + threadIdx.x;
    if (t < n) p[t] = 0.0f;
}

__global__ void degree_kernel(const int* __restrict__ dst, float* __restrict__ deg) {
    int e = blockIdx.x * blockDim.x + threadIdx.x;
    if (e >= N_EDGES) return;
    atomicAdd(&deg[dst[e]], 1.0f);
}

// in-place: deg -> D^{-1/2} with the self-loop (+1)
__global__ void dinv_kernel(float* __restrict__ deg) {
    int i = blockIdx.x * blockDim.x + threadIdx.x;
    if (i >= N_NODES) return;
    deg[i] = rsqrtf(deg[i] + 1.0f);
}

// x [N,14] fp32 -> act16 [N,64] f16, zero padded
__global__ void cvt_x_kernel(const float* __restrict__ x, _Float16* __restrict__ act) {
    int t = blockIdx.x * blockDim.x + threadIdx.x;
    if (t >= N_NODES * STRIDE) return;
    int i = t >> 6, f = t & 63;
    act[t] = (f < 14) ? (_Float16)x[i * 14 + f] : (_Float16)0.0f;
}

// ---------------------------------------------------------------- WMMA GEMM
// h[N,64(pad)] = act16[N,64(pad)] @ W[KIN,KOUT]  (f16 in, f32 accumulate)
// One wave per 16x16 tile; 4 waves/block share this block's N-tile of W,
// pre-swizzled in LDS into the exact per-lane B-fragment order so each lane
// fetches its fragment with two ds_load_b128.
//
// Fragment layouts (wave32, CDNA5 ISA 7.12.2):
//   A 16x32 f16 : lane = {hi=lane>>4, m=lane&15}, row M=m,
//                 halves j<8 -> K = kb+j, j>=8 -> K = kb+16+(j-8), kb=kc*32+hi*8
//   B 32x16 f16 : mirrored (lane column n=m, same K pattern)
//   D 16x16 f32 : lane column N=m, VGPR r -> row M = r + 8*hi
template <int KIN, int KOUT>
__global__ void gemm_wmma_kernel(const _Float16* __restrict__ act,
                                 const float* __restrict__ W,
                                 float* __restrict__ h) {
    constexpr int KCHUNKS = (KIN + 31) / 32;
    // WlB[idx]: idx = kc*512 + hi*256 + g*128 + n*8 + j
    //   -> element B[K = kc*32 + hi*8 + g*16 + j][col = ntile*16 + n]
    __shared__ __align__(16) _Float16 WlB[KCHUNKS * 512];

    const int tid = threadIdx.x;
    for (int idx = tid; idx < KCHUNKS * 512; idx += 128) {
        const int j  = idx & 7;
        const int n  = (idx >> 3) & 15;
        const int g  = (idx >> 7) & 1;
        const int hi = (idx >> 8) & 1;
        const int kc = idx >> 9;
        const int k   = kc * 32 + hi * 8 + g * 16 + j;
        const int col = blockIdx.y * 16 + n;
        WlB[idx] = (k < KIN && col < KOUT) ? (_Float16)W[k * KOUT + col]
                                           : (_Float16)0.0f;
    }
    __syncthreads();

    const int wave  = tid >> 5;
    const int lane  = tid & 31;
    const int mtile = blockIdx.x * 4 + wave;
    if (mtile >= M_TILES) return;           // wave-uniform exit

    const int hi = lane >> 4;
    const int m  = lane & 15;
    const int arow = mtile * 16 + m;        // exact tiling: always < N_NODES
    const int ncol = blockIdx.y * 16 + m;

    v8f c = {};
    #pragma unroll
    for (int kc = 0; kc < KCHUNKS; ++kc) {
        const _Float16* arp = act + (size_t)arow * STRIDE + kc * 32 + hi * 8;
        const v8h a_lo = *(const v8h*)arp;          // K = kb .. kb+7
        const v8h a_hi = *(const v8h*)(arp + 16);   // K = kb+16 .. kb+23
        const int bbase = kc * 512 + hi * 256 + m * 8;
        const v8h b_lo = *(const v8h*)&WlB[bbase];        // g = 0
        const v8h b_hi = *(const v8h*)&WlB[bbase + 128];  // g = 1
        const v16h a = __builtin_shufflevector(a_lo, a_hi,
            0,1,2,3,4,5,6,7,8,9,10,11,12,13,14,15);
        const v16h b = __builtin_shufflevector(b_lo, b_hi,
            0,1,2,3,4,5,6,7,8,9,10,11,12,13,14,15);
        c = __builtin_amdgcn_wmma_f32_16x16x32_f16(
                false, a, false, b, (short)0, c, false, false);
    }

    float* hrow = h + (size_t)(mtile * 16 + hi * 8) * STRIDE + ncol;
    #pragma unroll
    for (int r = 0; r < 8; ++r) {
        if (KOUT % 16 == 0 || ncol < KOUT)   // constant-folds except KOUT==2
            hrow[(size_t)r * STRIDE] = c[r];
    }
}

// ---------------------------------------------------------------- aggregation
// self-loop term initializes agg (no memset, no extra atomics):
// agg[i,f] = h[i,f] * dinv[i]^2
__global__ void selfloop_init_kernel(const float* __restrict__ h,
                                     const float* __restrict__ dinv,
                                     float* __restrict__ agg, int F) {
    int t = blockIdx.x * blockDim.x + threadIdx.x;
    if (t >= N_NODES * STRIDE) return;
    int i = t >> 6, f = t & 63;
    if (f < F) agg[t] = h[t] * dinv[i] * dinv[i];
}

// one thread per (edge, feature chunk): agg[dst] += h[src] * dinv[s]*dinv[d]
template <int LOGC, int WIDTH>
__global__ void edge_scatter_kernel(const float* __restrict__ h,
                                    float* __restrict__ agg,
                                    const float* __restrict__ dinv,
                                    const int* __restrict__ src,
                                    const int* __restrict__ dst,
                                    long long total) {
    long long t = (long long)blockIdx.x * blockDim.x + threadIdx.x;
    if (t >= total) return;
    const int e    = (int)(t >> LOGC);
    const int cidx = (int)(t & ((1 << LOGC) - 1));
    const int f0   = cidx * 4;
    const int s = src[e];
    const int d = dst[e];
    const float nrm = dinv[s] * dinv[d];
    const float* hp = h   + (size_t)s * STRIDE + f0;
    float*       ap = agg + (size_t)d * STRIDE + f0;
    if constexpr (WIDTH == 4) {
        const float4 v = *(const float4*)hp;      // 16B aligned (f0 % 4 == 0)
        atomicAdd(ap + 0, v.x * nrm);
        atomicAdd(ap + 1, v.y * nrm);
        atomicAdd(ap + 2, v.z * nrm);
        atomicAdd(ap + 3, v.w * nrm);
    } else {
        const float2 v = *(const float2*)hp;
        atomicAdd(ap + 0, v.x * nrm);
        atomicAdd(ap + 1, v.y * nrm);
    }
}

// bias + ReLU + f16 convert (feeds next layer's GEMM); pads with zeros
__global__ void finish_relu_cvt_kernel(const float* __restrict__ agg,
                                       const float* __restrict__ bias,
                                       _Float16* __restrict__ act, int F) {
    int t = blockIdx.x * blockDim.x + threadIdx.x;
    if (t >= N_NODES * STRIDE) return;
    int f = t & 63;
    float v = 0.0f;
    if (f < F) {
        v = agg[t] + bias[f];
        v = v > 0.0f ? v : 0.0f;
    }
    act[t] = (_Float16)v;
}

// last layer: compact [N,2] fp32 = agg + b4  (no relu)
__global__ void final_bias_kernel(const float* __restrict__ agg,
                                  const float* __restrict__ b4,
                                  float* __restrict__ hfin) {
    int t = blockIdx.x * blockDim.x + threadIdx.x;
    if (t >= N_NODES * 2) return;
    int i = t >> 1, f = t & 1;
    hfin[t] = agg[i * STRIDE + f] + b4[f];
}

// ---------------------------------------------------------------- pooling
__global__ void pool_init_kernel(float* pooled) {
    int t = blockIdx.x * blockDim.x + threadIdx.x;
    if (t < N_GRAPHS * 2) pooled[t] = __int_as_float(0xff800000); // -inf
}

__device__ __forceinline__ void atomicMaxF(float* addr, float val) {
    int* ai = (int*)addr;
    int old = __float_as_int(*addr);
    while (__int_as_float(old) < val) {
        int assumed = old;
        old = atomicCAS(ai, assumed, __float_as_int(val));
        if (old == assumed) break;
    }
}

__global__ void pool_max_kernel(const float* __restrict__ hfin,
                                const int* __restrict__ batch,
                                float* __restrict__ pooled) {
    int t = blockIdx.x * blockDim.x + threadIdx.x;
    if (t >= N_NODES * 2) return;
    int i = t >> 1, c = t & 1;
    atomicMaxF(&pooled[batch[i] * 2 + c], hfin[t]);
}

__global__ void logsoftmax_kernel(const float* __restrict__ pooled,
                                  float* __restrict__ out) {
    int g = blockIdx.x * blockDim.x + threadIdx.x;
    if (g >= N_GRAPHS) return;
    float p0 = pooled[2 * g + 0];
    float p1 = pooled[2 * g + 1];
    // guard non-finite (empty graphs) -> 0, matching the reference
    if ((__float_as_uint(p0) & 0x7f800000u) == 0x7f800000u) p0 = 0.0f;
    if ((__float_as_uint(p1) & 0x7f800000u) == 0x7f800000u) p1 = 0.0f;
    float mx = fmaxf(p0, p1);
    float l  = mx + logf(expf(p0 - mx) + expf(p1 - mx));
    out[2 * g + 0] = p0 - l;
    out[2 * g + 1] = p1 - l;
}

// ---------------------------------------------------------------- launch

static inline size_t align256(size_t x) { return (x + 255) & ~(size_t)255; }

extern "C" void kernel_launch(void* const* d_in, const int* in_sizes, int n_in,
                              void* d_out, int out_size, void* d_ws, size_t ws_size,
                              hipStream_t stream) {
    const float* x     = (const float*)d_in[0];
    const int*   ei    = (const int*)d_in[1];     // [2, E]
    const int*   batch = (const int*)d_in[2];
    const float* W1 = (const float*)d_in[3];
    const float* b1 = (const float*)d_in[4];
    const float* W2 = (const float*)d_in[5];
    const float* b2 = (const float*)d_in[6];
    const float* W3 = (const float*)d_in[7];
    const float* b3 = (const float*)d_in[8];
    const float* W4 = (const float*)d_in[9];
    const float* b4 = (const float*)d_in[10];
    float* out = (float*)d_out;

    const int* src = ei;
    const int* dst = ei + N_EDGES;

    // workspace carve-up (~64.5 MB)
    char* p = (char*)d_ws;
    float*    dinv   = (float*)p;    p += align256((size_t)N_NODES * 4);
    _Float16* act16  = (_Float16*)p; p += align256((size_t)N_NODES * STRIDE * 2);
    float*    h      = (float*)p;    p += align256((size_t)N_NODES * STRIDE * 4);
    float*    agg    = (float*)p;    p += align256((size_t)N_NODES * STRIDE * 4);
    float*    pooled = (float*)p;    p += align256((size_t)N_GRAPHS * 2 * 4);
    (void)ws_size; (void)n_in; (void)in_sizes; (void)out_size;

    const int TB  = 256;
    const int nNS = (N_NODES * STRIDE + TB - 1) / TB;
    const int mblocks = (M_TILES + 3) / 4;    // 4 waves per block

    // degree -> dinv
    zero_f32_kernel<<<(N_NODES + TB - 1) / TB, TB, 0, stream>>>(dinv, N_NODES);
    degree_kernel<<<(N_EDGES + TB - 1) / TB, TB, 0, stream>>>(dst, dinv);
    dinv_kernel<<<(N_NODES + TB - 1) / TB, TB, 0, stream>>>(dinv);

    // layer 0 input
    cvt_x_kernel<<<nNS, TB, 0, stream>>>(x, act16);

    const long long tot64 = (long long)N_EDGES << 4;   // F=64: 16 chunks/edge
    const long long tot32 = (long long)N_EDGES << 3;   // F=32:  8 chunks/edge
    const long long tot2  = (long long)N_EDGES;        // F=2:   1 chunk/edge

    // ---- layer 1: 14 -> 64, relu
    gemm_wmma_kernel<14, 64><<<dim3(mblocks, 4), 128, 0, stream>>>(act16, W1, h);
    selfloop_init_kernel<<<nNS, TB, 0, stream>>>(h, dinv, agg, 64);
    edge_scatter_kernel<4, 4><<<(unsigned)((tot64 + TB - 1) / TB), TB, 0, stream>>>(
        h, agg, dinv, src, dst, tot64);
    finish_relu_cvt_kernel<<<nNS, TB, 0, stream>>>(agg, b1, act16, 64);

    // ---- layer 2: 64 -> 64, relu
    gemm_wmma_kernel<64, 64><<<dim3(mblocks, 4), 128, 0, stream>>>(act16, W2, h);
    selfloop_init_kernel<<<nNS, TB, 0, stream>>>(h, dinv, agg, 64);
    edge_scatter_kernel<4, 4><<<(unsigned)((tot64 + TB - 1) / TB), TB, 0, stream>>>(
        h, agg, dinv, src, dst, tot64);
    finish_relu_cvt_kernel<<<nNS, TB, 0, stream>>>(agg, b2, act16, 64);

    // ---- layer 3: 64 -> 32, relu
    gemm_wmma_kernel<64, 32><<<dim3(mblocks, 2), 128, 0, stream>>>(act16, W3, h);
    selfloop_init_kernel<<<nNS, TB, 0, stream>>>(h, dinv, agg, 32);
    edge_scatter_kernel<3, 4><<<(unsigned)((tot32 + TB - 1) / TB), TB, 0, stream>>>(
        h, agg, dinv, src, dst, tot32);
    finish_relu_cvt_kernel<<<nNS, TB, 0, stream>>>(agg, b3, act16, 32);

    // ---- layer 4: 32 -> 2, no relu
    gemm_wmma_kernel<32, 2><<<dim3(mblocks, 1), 128, 0, stream>>>(act16, W4, h);
    selfloop_init_kernel<<<nNS, TB, 0, stream>>>(h, dinv, agg, 2);
    edge_scatter_kernel<0, 2><<<(unsigned)((tot2 + TB - 1) / TB), TB, 0, stream>>>(
        h, agg, dinv, src, dst, tot2);
    final_bias_kernel<<<(N_NODES * 2 + TB - 1) / TB, TB, 0, stream>>>(agg, b4, h);

    // max pool + log_softmax
    pool_init_kernel<<<(N_GRAPHS * 2 + TB - 1) / TB, TB, 0, stream>>>(pooled);
    pool_max_kernel<<<(N_NODES * 2 + TB - 1) / TB, TB, 0, stream>>>(h, batch, pooled);
    logsoftmax_kernel<<<(N_GRAPHS + TB - 1) / TB, TB, 0, stream>>>(pooled, out);
}
